// iSHTCUDA_75660143886828
// MI455X (gfx1250) — matrix operand order repaired and checked by reading, not compiled, not run on hardware
//
#include <hip/hip_runtime.h>

// ---------------------------------------------------------------------------
// Inverse SHT on HEALPix (NSIDE=128) for MI455X / gfx1250.
//
// Stage 1: Xr/Xi[b,k,m] = sum_l x[b,l,m,{re,im}] * pct[m,l,k]
//          per-m GEMM [64 x 511] x [511 x 511], bf16 WMMA, f32 accum
// Stage 2: grid[b,k,j]  = sum_m Xr[b,k,m]*A[k,j,m] + Xi[b,k,m]*B[k,j,m]
//          per-ring GEMM [64 x 257] x [257 x 512], bf16 WMMA, f32 accum
// Stage 3: out[b,p] = grid[b, fidx[p]]
//
// LDS panels are stored in WMMA *fragment order* (each lane's 16 bf16
// contiguous -> ds_load_b128 pairs), double-buffered so global staging for
// step i+1 overlaps the WMMAs of step i. Tail guards are hoisted into
// uniform fast/slow paths so the hot loop is branch-free.
// ---------------------------------------------------------------------------

#define NLAT_   511
#define NLON_   512
#define LMAX_   511
#define MMAX_   257
#define NPIX_   196608
#define NBATCH_ 64

typedef __attribute__((ext_vector_type(16))) __bf16 v16bf;
typedef __attribute__((ext_vector_type(8)))  __bf16 v8bf;
typedef __attribute__((ext_vector_type(8)))  float  v8f;

struct __align__(8) f2 { float x, y; };

#define X_BSTR ((size_t)LMAX_ * MMAX_ * 2)   // x batch stride (floats)
#define X_LSTR ((size_t)MMAX_ * 2)           // x l stride (floats)

// Fragment-order mapping (16-bit A operand, 16x32):
//   element e of lane (half*16 + row) holds K = (e>>3)*16 + half*8 + (e&7)
//   => a 16-long K-run {g*16 .. g*16+15} for one row is two contiguous
//      8-element chunks at [row][g*8] (K%16<8) and [row+16][g*8] (K%16>=8).
// B operand (32x16): element e of lane L holds K=(e>>3)*16+(L>>4)*8+(e&7), N=L&15;
//   storage bP[((K>>3)&1)*16 + N][(K>>4)*8 + (K&7)] reads back as exactly bP[L][0..15].

// ---------------------------------------------------------------------------
// Stage 1: Legendre contraction.  grid.x = MMAX_*32, block = 128 (4 waves).
// ---------------------------------------------------------------------------
__global__ __launch_bounds__(128) void sht_legendre_wmma(
    const float* __restrict__ x, const float* __restrict__ pct,
    float* __restrict__ Xr, float* __restrict__ Xi)
{
    __shared__ __align__(32) __bf16 aR[2][4][32][16];   // 8 KB
    __shared__ __align__(32) __bf16 aI[2][4][32][16];   // 8 KB
    __shared__ __align__(32) __bf16 bP[2][32][16];      // 2 KB

    const int m    = blockIdx.x >> 5;
    const int k0   = (blockIdx.x & 31) * 16;
    const int tid  = threadIdx.x;
    const int wave = tid >> 5;
    const int lane = tid & 31;
    const int half = lane >> 4;
    const int col  = lane & 15;

    const float* pbase = pct + (size_t)m * LMAX_ * NLAT_;

    // x-panel staging task: thread -> (batch row, 16-long l-run)
    const int xb = tid >> 1;          // 0..63
    const int xg = tid & 1;           // which 16-run of the 32-l panel
    const int xw = xb >> 4, xr = xb & 15;
    const float* xrow = x + (size_t)xb * X_BSTR + (size_t)m * 2;

    // pct-panel staging task: thread -> (l row, 4-long k-run)
    const int pl = tid >> 2;          // 0..31
    const int pq = tid & 3;

    auto fill = [&](int l0, int buf) {
        // ---- x panel (A operands, Re+Im), strided float2 loads ----
        const int lb = l0 + xg * 16;
        float fr[16], fi[16];
        if (lb + 15 < LMAX_) {
#pragma unroll
            for (int t = 0; t < 16; ++t) {
                f2 v = *(const f2*)(xrow + (size_t)(lb + t) * X_LSTR);
                fr[t] = v.x; fi[t] = v.y;
            }
        } else {
#pragma unroll
            for (int t = 0; t < 16; ++t) {
                float re = 0.0f, im = 0.0f;
                if (lb + t < LMAX_) {
                    f2 v = *(const f2*)(xrow + (size_t)(lb + t) * X_LSTR);
                    re = v.x; im = v.y;
                }
                fr[t] = re; fi[t] = im;
            }
        }
        v8bf c0r, c1r, c0i, c1i;
#pragma unroll
        for (int t = 0; t < 8; ++t) {
            c0r[t] = (__bf16)fr[t];     c0i[t] = (__bf16)fi[t];
            c1r[t] = (__bf16)fr[t + 8]; c1i[t] = (__bf16)fi[t + 8];
        }
        *(v8bf*)&aR[buf][xw][xr][xg * 8]      = c0r;   // ds_store_b128
        *(v8bf*)&aR[buf][xw][xr + 16][xg * 8] = c1r;
        *(v8bf*)&aI[buf][xw][xr][xg * 8]      = c0i;
        *(v8bf*)&aI[buf][xw][xr + 16][xg * 8] = c1i;

        // ---- pct panel (B operand), k-contiguous loads, b16 scatter ----
        {
            const int l = l0 + pl;
            float pv[4] = {0.f, 0.f, 0.f, 0.f};
            if (l < LMAX_) {
                const float* pp = pbase + (size_t)l * NLAT_ + k0 + pq * 4;
                if (k0 + 15 < NLAT_) {
#pragma unroll
                    for (int n = 0; n < 4; ++n) pv[n] = pp[n];
                } else {
#pragma unroll
                    for (int n = 0; n < 4; ++n)
                        if (k0 + pq * 4 + n < NLAT_) pv[n] = pp[n];
                }
                // prefetch the panel two steps ahead
                if (l0 + 32 + pl < LMAX_)
                    __builtin_prefetch(pbase + (size_t)(l0 + 32 + pl) * NLAT_ + k0, 0, 0);
            }
            const int lhi = ((pl >> 3) & 1) << 4;
            const int e   = (pl & 7) | ((pl >> 4) << 3);
#pragma unroll
            for (int n = 0; n < 4; ++n)
                bP[buf][lhi + pq * 4 + n][e] = (__bf16)pv[n];
        }
    };

    v8f accR = {};
    v8f accI = {};
    int p = 0;
    fill(0, 0);
    __syncthreads();

    for (int it = 0; it < 16; ++it) {               // 16 * 32 covers L=511
        v16bf fAR = *(const v16bf*)&aR[p][wave][lane][0];   // 2x ds_load_b128
        v16bf fAI = *(const v16bf*)&aI[p][wave][lane][0];
        v16bf fB  = *(const v16bf*)&bP[p][lane][0];
        if (it + 1 < 16) fill((it + 1) * 32, p ^ 1);
        accR = __builtin_amdgcn_wmma_f32_16x16x32_bf16(
            false, fAR, false, fB, (short)0, accR, false, false);
        accI = __builtin_amdgcn_wmma_f32_16x16x32_bf16(
            false, fAI, false, fB, (short)0, accI, false, false);
        __syncthreads();
        p ^= 1;
    }

    // C/D layout: VGPR r -> M = r + half*8, N = lane&15.
    const int k = k0 + col;
    if (k < NLAT_) {
#pragma unroll
        for (int r = 0; r < 8; ++r) {
            int b = (wave << 4) + (half << 3) + r;
            size_t o = (size_t)b * NLAT_ * MMAX_ + (size_t)k * MMAX_ + m;
            Xr[o] = accR[r];
            Xi[o] = accI[r];
        }
    }
}

// ---------------------------------------------------------------------------
// Stage 2: per-ring synthesis.  grid.x = NLAT_*32, block = 128 (4 waves).
// ---------------------------------------------------------------------------
__global__ __launch_bounds__(128) void sht_synth_wmma(
    const float* __restrict__ Xr, const float* __restrict__ Xi,
    const float* __restrict__ Atab, const float* __restrict__ Btab,
    float* __restrict__ grid)
{
    __shared__ __align__(32) __bf16 aR[2][4][32][16];
    __shared__ __align__(32) __bf16 aI[2][4][32][16];
    __shared__ __align__(32) __bf16 bA[2][32][16];
    __shared__ __align__(32) __bf16 bB[2][32][16];

    const int k    = blockIdx.x >> 5;          // ring 0..510
    const int j0   = (blockIdx.x & 31) * 16;
    const int tid  = threadIdx.x;
    const int wave = tid >> 5;
    const int lane = tid & 31;
    const int half = lane >> 4;
    const int col  = lane & 15;

    // Xr/Xi staging task: thread -> (batch row, 16-long m-run), m contiguous
    const int xb = tid >> 1;
    const int xg = tid & 1;
    const int xw = xb >> 4, xr = xb & 15;
    const float* xrrow = Xr + (size_t)xb * NLAT_ * MMAX_ + (size_t)k * MMAX_;
    const float* xirow = Xi + (size_t)xb * NLAT_ * MMAX_ + (size_t)k * MMAX_;

    // table staging task: thread -> (table, j row, 8-long m-run)
    const int ttab = tid >> 6;                 // 0 = A, 1 = B
    const int trr  = tid & 63;
    const int tN   = trr >> 2;
    const int tq   = trr & 3;
    const float* tbase = (ttab ? Btab : Atab)
                         + ((size_t)k * NLON_ + j0 + tN) * MMAX_ + tq * 8;
    const int tlane = ((tq & 1) << 4) + tN;    // ((m_local>>3)&1 ? 16 : 0) + N
    const int te0   = (tq >> 1) << 3;          // (m_local>=16 ? 8 : 0)

    auto fill = [&](int m0, int buf) {
        // ---- Xr/Xi panel (A operands), contiguous along m ----
        const int mb = m0 + xg * 16;
        float fr[16], fi[16];
        if (mb + 15 < MMAX_) {
#pragma unroll
            for (int t = 0; t < 16; ++t) { fr[t] = xrrow[mb + t]; fi[t] = xirow[mb + t]; }
        } else {
#pragma unroll
            for (int t = 0; t < 16; ++t) {
                bool ok = (mb + t) < MMAX_;
                fr[t] = ok ? xrrow[mb + t] : 0.0f;
                fi[t] = ok ? xirow[mb + t] : 0.0f;
            }
        }
        v8bf c0r, c1r, c0i, c1i;
#pragma unroll
        for (int t = 0; t < 8; ++t) {
            c0r[t] = (__bf16)fr[t];     c0i[t] = (__bf16)fi[t];
            c1r[t] = (__bf16)fr[t + 8]; c1i[t] = (__bf16)fi[t + 8];
        }
        *(v8bf*)&aR[buf][xw][xr][xg * 8]      = c0r;
        *(v8bf*)&aR[buf][xw][xr + 16][xg * 8] = c1r;
        *(v8bf*)&aI[buf][xw][xr][xg * 8]      = c0i;
        *(v8bf*)&aI[buf][xw][xr + 16][xg * 8] = c1i;

        // ---- A/B tables (B operands), contiguous along m, b128 LDS store ----
        {
            float ft[8];
            const int mtb = m0 + tq * 8;
            if (mtb + 7 < MMAX_) {
#pragma unroll
                for (int t = 0; t < 8; ++t) ft[t] = tbase[m0 + t];
            } else {
#pragma unroll
                for (int t = 0; t < 8; ++t)
                    ft[t] = (mtb + t) < MMAX_ ? tbase[m0 + t] : 0.0f;
            }
            v8bf ct;
#pragma unroll
            for (int t = 0; t < 8; ++t) ct[t] = (__bf16)ft[t];
            if (ttab) *(v8bf*)&bB[buf][tlane][te0] = ct;
            else      *(v8bf*)&bA[buf][tlane][te0] = ct;
        }
    };

    v8f acc = {};
    int p = 0;
    fill(0, 0);
    __syncthreads();

    for (int it = 0; it < 9; ++it) {            // 9 * 32 covers M=257
        v16bf fAR = *(const v16bf*)&aR[p][wave][lane][0];
        v16bf fAI = *(const v16bf*)&aI[p][wave][lane][0];
        v16bf fBA = *(const v16bf*)&bA[p][lane][0];
        v16bf fBB = *(const v16bf*)&bB[p][lane][0];
        if (it + 1 < 9) fill((it + 1) * 32, p ^ 1);
        acc = __builtin_amdgcn_wmma_f32_16x16x32_bf16(
            false, fAR, false, fBA, (short)0, acc, false, false);
        acc = __builtin_amdgcn_wmma_f32_16x16x32_bf16(
            false, fAI, false, fBB, (short)0, acc, false, false);
        __syncthreads();
        p ^= 1;
    }

    const int j = j0 + col;                     // NLON_=512: always in range
#pragma unroll
    for (int r = 0; r < 8; ++r) {
        int b = (wave << 4) + (half << 3) + r;
        grid[(size_t)b * ((size_t)NLAT_ * NLON_) + (size_t)k * NLON_ + j] = acc[r];
    }
}

// ---------------------------------------------------------------------------
// Stage 3: RING-order gather; one fidx load per pixel reused over 64 batches.
// ---------------------------------------------------------------------------
__global__ __launch_bounds__(256) void sht_gather(
    const float* __restrict__ grid, const int* __restrict__ fidx,
    float* __restrict__ out)
{
    int p = blockIdx.x * 256 + threadIdx.x;
    if (p >= NPIX_) return;
    int f = fidx[p];
#pragma unroll 4
    for (int b = 0; b < NBATCH_; ++b) {
        out[(size_t)b * NPIX_ + p] = grid[(size_t)b * ((size_t)NLAT_ * NLON_) + f];
    }
}

// ---------------------------------------------------------------------------
extern "C" void kernel_launch(void* const* d_in, const int* in_sizes, int n_in,
                              void* d_out, int out_size, void* d_ws, size_t ws_size,
                              hipStream_t stream)
{
    const float* x    = (const float*)d_in[0];   // [4,16,511,257,2]
    const float* pct  = (const float*)d_in[1];   // [257,511,511]
    const float* Atab = (const float*)d_in[2];   // [511,512,257]
    const float* Btab = (const float*)d_in[3];   // [511,512,257]
    const int*   fidx = (const int*)d_in[4];     // [196608]
    float* out = (float*)d_out;                  // [64,196608]

    // Workspace: Xr (8,404,928 f32) | Xi (8,404,928) | grid (16,744,448)
    float* ws = (float*)d_ws;
    const size_t XSZ = (size_t)NBATCH_ * NLAT_ * MMAX_;
    float* Xr   = ws;
    float* Xi   = ws + XSZ;
    float* grd  = ws + 2 * XSZ;

    sht_legendre_wmma<<<dim3(MMAX_ * 32), dim3(128), 0, stream>>>(x, pct, Xr, Xi);
    sht_synth_wmma<<<dim3(NLAT_ * 32), dim3(128), 0, stream>>>(Xr, Xi, Atab, Btab, grd);
    sht_gather<<<dim3((NPIX_ + 255) / 256), dim3(256), 0, stream>>>(grd, fidx, out);
}
